// GPT_27247272526006
// MI455X (gfx1250) — matrix operand order repaired and checked by reading, not compiled, not run on hardware
//
#include <hip/hip_runtime.h>
#include <math.h>

// ---------------- model constants ----------------
#define VSZ 50304
#define DM  768
#define NH  12
#define HD  64
#define NL  12
#define BB  2
#define SS  1024
#define TT  (BB * SS)         // 2048 tokens
#define D3  (3 * DM)          // 2304
#define D4  (4 * DM)          // 3072

typedef __bf16 v16bf __attribute__((ext_vector_type(16)));
typedef float  v8f   __attribute__((ext_vector_type(8)));

union Frag { v16bf v; unsigned short u[16]; };

__device__ __forceinline__ unsigned short f2bf(float f) {
    union { float f; unsigned int u; } c; c.f = f;
    unsigned int u = c.u;
    return (unsigned short)((u + 0x7FFFu + ((u >> 16) & 1u)) >> 16);
}

__device__ __forceinline__ v8f wmma_bf16(const Frag& a, const Frag& b, v8f c) {
    return __builtin_amdgcn_wmma_f32_16x16x32_bf16(
        false, a.v, false, b.v, (short)0, c, false, false);
}

// 16-byte async DMA: global -> LDS, tracked by ASYNCcnt (gfx1250 path).
__device__ __forceinline__ void async_ld16(void* lds, const void* g) {
    unsigned l = (unsigned)(unsigned long long)lds;        // LDS byte offset
    unsigned long long ga = (unsigned long long)g;
    asm volatile("global_load_async_to_lds_b128 %0, %1, off"
                 :: "v"(l), "v"(ga) : "memory");
}
__device__ __forceinline__ void wait_async() {
    asm volatile("s_wait_asynccnt 0" ::: "memory");
}

// ---------------- embedding: x = wte[ids] + wpe[pos] ----------------
__global__ __launch_bounds__(256) void embed_kernel(
    const int* __restrict__ ids, const float* __restrict__ wte,
    const float* __restrict__ wpe, float* __restrict__ x) {
    int t = blockIdx.x;
    int id = ids[t];
    int pos = t % SS;
    for (int d = threadIdx.x; d < DM; d += 256)
        x[t * DM + d] = wte[id * DM + d] + wpe[pos * DM + d];
}

// ---------------- layernorm: fp32 in -> bf16 out ----------------
__global__ __launch_bounds__(256) void ln_kernel(
    const float* __restrict__ x, const float* __restrict__ g,
    const float* __restrict__ b, unsigned short* __restrict__ y, float eps) {
    __shared__ float s1[256], s2[256];
    int t = blockIdx.x, tid = threadIdx.x;
    float ls = 0.f, lq = 0.f;
    for (int d = tid; d < DM; d += 256) {
        float v = x[t * DM + d]; ls += v; lq += v * v;
    }
    s1[tid] = ls; s2[tid] = lq; __syncthreads();
    for (int s = 128; s > 0; s >>= 1) {
        if (tid < s) { s1[tid] += s1[tid + s]; s2[tid] += s2[tid + s]; }
        __syncthreads();
    }
    float mean = s1[0] * (1.f / DM);
    float var  = s2[0] * (1.f / DM) - mean * mean;
    float rstd = rsqrtf(var + eps);
    for (int d = tid; d < DM; d += 256) {
        float v = x[t * DM + d];
        y[t * DM + d] = f2bf((v - mean) * rstd * g[d] + b[d]);
    }
}

// -------- WMMA GEMM: C[M,N] = A_bf16[M,K] @ B_f32 + bias (+epilogue) -------
// EPI: 0 = bias; 1 = bias + residual add; 2 = bias + exact GELU.
// BT : false -> B row-major [K,N]; true -> B is [N,K] (use B^T).
// OUTBF: write bf16 (true) or fp32 (false).
template <int EPI, bool BT, bool OUTBF>
__global__ __launch_bounds__(256) void gemm_kernel(
    const unsigned short* __restrict__ A, const float* __restrict__ B,
    const float* __restrict__ bias, const float* __restrict__ res,
    void* __restrict__ Cout, int M, int N, int K) {
    __shared__ unsigned short As[128][34];   // +2 pad: conflict-free frag reads
    __shared__ unsigned short Bs[32][130];

    const int t    = threadIdx.x;
    const int lane = t & 31;
    const int wid  = t >> 5;                 // 8 waves
    const int wm   = wid >> 1;               // 0..3 -> 32-row slice
    const int wn   = wid & 1;                // 0..1 -> 64-col slice
    const int rw   = wm * 32, cw = wn * 64;
    const int n0   = blockIdx.x * 128;
    const int m0   = blockIdx.y * 128;

    v8f acc[2][4];
    #pragma unroll
    for (int mi = 0; mi < 2; ++mi)
        #pragma unroll
        for (int ni = 0; ni < 4; ++ni)
            #pragma unroll
            for (int r = 0; r < 8; ++r) acc[mi][ni][r] = 0.f;

    const int nk = K >> 5;
    for (int kt = 0; kt < nk; ++kt) {
        __syncthreads();
        // stage A tile 128x32 bf16 via async DMA to LDS (4 x 16B per row)
        #pragma unroll
        for (int it = 0; it < 2; ++it) {
            int c = t + it * 256;            // 512 16-byte chunks
            int row = c >> 2, ch = c & 3;
            async_ld16(&As[row][ch * 8],
                       &A[(size_t)(m0 + row) * K + kt * 32 + ch * 8]);
        }
        // stage B tile 32x128 (fp32 -> bf16 convert in flight)
        if constexpr (!BT) {
            int row = t >> 5, nq = (t & 31) << 2;
            #pragma unroll
            for (int it = 0; it < 4; ++it, row += 8) {
                float4 b4 = *(const float4*)&B[(size_t)(kt * 32 + row) * N + n0 + nq];
                Bs[row][nq + 0] = f2bf(b4.x); Bs[row][nq + 1] = f2bf(b4.y);
                Bs[row][nq + 2] = f2bf(b4.z); Bs[row][nq + 3] = f2bf(b4.w);
            }
        } else {
            int nn = t >> 3, kq = (t & 7) << 2;
            #pragma unroll
            for (int it = 0; it < 4; ++it, nn += 32) {
                float4 b4 = *(const float4*)&B[(size_t)(n0 + nn) * K + kt * 32 + kq];
                Bs[kq + 0][nn] = f2bf(b4.x); Bs[kq + 1][nn] = f2bf(b4.y);
                Bs[kq + 2][nn] = f2bf(b4.z); Bs[kq + 3][nn] = f2bf(b4.w);
            }
        }
        if (kt + 1 < nk) {   // cacheline hint for next weight tile
            if constexpr (!BT)
                __builtin_prefetch(&B[(size_t)((kt + 1) * 32 + (t >> 5)) * N + n0], 0, 1);
            else
                __builtin_prefetch(&B[(size_t)(n0 + (t >> 3)) * K + (kt + 1) * 32], 0, 1);
        }
        wait_async();
        __syncthreads();

        Frag aF[2], bF[4];
        #pragma unroll
        for (int mi = 0; mi < 2; ++mi) {
            int m = rw + mi * 16 + (lane & 15);
            #pragma unroll
            for (int i = 0; i < 16; ++i) {
                int k = ((i >> 3) << 4) + ((lane >> 4) << 3) + (i & 7);
                aF[mi].u[i] = As[m][k];
            }
        }
        #pragma unroll
        for (int ni = 0; ni < 4; ++ni)
            #pragma unroll
            for (int j = 0; j < 16; ++j)
                bF[ni].u[j] = Bs[((j >> 3) << 4) + ((lane >> 4) << 3) + (j & 7)][cw + ni * 16 + (lane & 15)];

        #pragma unroll
        for (int mi = 0; mi < 2; ++mi)
            #pragma unroll
            for (int ni = 0; ni < 4; ++ni)
                acc[mi][ni] = wmma_bf16(aF[mi], bF[ni], acc[mi][ni]);
    }

    // epilogue: C-layout element (m = r + 8*(lane>>4), n = lane&15)
    #pragma unroll
    for (int mi = 0; mi < 2; ++mi)
        #pragma unroll
        for (int ni = 0; ni < 4; ++ni)
            #pragma unroll
            for (int r = 0; r < 8; ++r) {
                int gm = m0 + rw + mi * 16 + r + ((lane >> 4) << 3);
                int gn = n0 + cw + ni * 16 + (lane & 15);
                float v = acc[mi][ni][r] + bias[gn];
                if constexpr (EPI == 1) v += res[(size_t)gm * N + gn];
                if constexpr (EPI == 2) v = 0.5f * v * (1.f + erff(v * 0.70710678118f));
                if constexpr (OUTBF)
                    ((unsigned short*)Cout)[(size_t)gm * N + gn] = f2bf(v);
                else
                    ((float*)Cout)[(size_t)gm * N + gn] = v;
            }
}

// ---------------- flash attention (causal), WMMA QK^T and P·V --------------
// grid: (S/64, NH, BB); block: 128 (4 waves, each owns 16 query rows)
__global__ __launch_bounds__(128) void attn_kernel(
    const unsigned short* __restrict__ qkv,  // [TT, 3D] bf16
    const float* __restrict__ amask,         // [BB, SS]
    unsigned short* __restrict__ out) {      // [TT, DM] bf16
    __shared__ unsigned short KtT[64][66];     // [hd][key] (transposed)
    __shared__ unsigned short Vt[64][66];      // [key][hd]
    __shared__ unsigned short Pw[4][16][66];   // per-wave P tile

    const int qt = blockIdx.x, h = blockIdx.y, b = blockIdx.z;
    const int tid = threadIdx.x, lane = tid & 31, wid = tid >> 5;
    const int qrow0 = qt * 64 + wid * 16;

    // Q fragments (bf16 direct; 1/sqrt(HD) applied to scores later)
    Frag qF[2];
    {
        int m = lane & 15;
        size_t base = (size_t)(b * SS + qrow0 + m) * D3 + h * HD;
        #pragma unroll
        for (int f = 0; f < 2; ++f)
            #pragma unroll
            for (int i = 0; i < 16; ++i) {
                int k = ((i >> 3) << 4) + ((lane >> 4) << 3) + (i & 7);
                qF[f].u[i] = qkv[base + f * 32 + k];
            }
    }

    float rm[8], rs[8];
    v8f o[4];
    #pragma unroll
    for (int r = 0; r < 8; ++r) { rm[r] = -1e30f; rs[r] = 0.f; }
    #pragma unroll
    for (int nt = 0; nt < 4; ++nt)
        #pragma unroll
        for (int r = 0; r < 8; ++r) o[nt][r] = 0.f;

    for (int kt = 0; kt <= qt; ++kt) {
        __syncthreads();
        // V tile: 64 rows x 128B -> 512 16B chunks via async DMA
        #pragma unroll
        for (int it = 0; it < 4; ++it) {
            int c = tid + it * 128;          // 512 chunks
            int row = c >> 3, ch = c & 7;
            async_ld16(&Vt[row][ch * 8],
                       &qkv[(size_t)(b * SS + kt * 64 + row) * D3 + 2 * DM + h * HD + ch * 8]);
        }
        // K tile needs transpose: elementwise bf16 copy
        {
            int kk = tid >> 1, ds = (tid & 1) * 32;
            size_t base = (size_t)(b * SS + kt * 64 + kk) * D3 + DM + h * HD + ds;
            #pragma unroll
            for (int j = 0; j < 32; ++j)
                KtT[ds + j][kk] = qkv[base + j];
        }
        wait_async();
        __syncthreads();

        // scores S = Q K^T (contraction over hd = 64 -> two wmma per tile)
        v8f sf[4];
        #pragma unroll
        for (int nt = 0; nt < 4; ++nt) {
            v8f c;
            #pragma unroll
            for (int r = 0; r < 8; ++r) c[r] = 0.f;
            Frag b0, b1;
            #pragma unroll
            for (int j = 0; j < 16; ++j) {
                int k = ((j >> 3) << 4) + ((lane >> 4) << 3) + (j & 7);
                b0.u[j] = KtT[k][nt * 16 + (lane & 15)];
                b1.u[j] = KtT[32 + k][nt * 16 + (lane & 15)];
            }
            c = wmma_bf16(qF[0], b0, c);
            c = wmma_bf16(qF[1], b1, c);
            sf[nt] = c;
        }

        // scale + causal + padding mask
        float madd[4];
        #pragma unroll
        for (int nt = 0; nt < 4; ++nt) {
            int gk = kt * 64 + nt * 16 + (lane & 15);
            madd[nt] = (1.f - amask[b * SS + gk]) * -10000.f;
        }
        #pragma unroll
        for (int nt = 0; nt < 4; ++nt)
            #pragma unroll
            for (int r = 0; r < 8; ++r) {
                int gq = qrow0 + r + ((lane >> 4) << 3);
                int gk = kt * 64 + nt * 16 + (lane & 15);
                float s = sf[nt][r] * 0.125f;       // 1/sqrt(64)
                if (gk > gq) s = -10000.f;
                sf[nt][r] = s + madd[nt];
            }

        // online softmax per row (each row spread over the 16 lanes of one half)
        #pragma unroll
        for (int r = 0; r < 8; ++r) {
            float mx = sf[0][r];
            #pragma unroll
            for (int nt = 1; nt < 4; ++nt) mx = fmaxf(mx, sf[nt][r]);
            #pragma unroll
            for (int xm = 1; xm < 16; xm <<= 1) mx = fmaxf(mx, __shfl_xor(mx, xm, 32));
            float nm = fmaxf(rm[r], mx);
            float corr = __expf(rm[r] - nm);
            rm[r] = nm; rs[r] *= corr;
            #pragma unroll
            for (int nt = 0; nt < 4; ++nt) o[nt][r] *= corr;
            float lsum = 0.f;
            #pragma unroll
            for (int nt = 0; nt < 4; ++nt) {
                float p = __expf(sf[nt][r] - nm);
                sf[nt][r] = p; lsum += p;
            }
            #pragma unroll
            for (int xm = 1; xm < 16; xm <<= 1) lsum += __shfl_xor(lsum, xm, 32);
            rs[r] += lsum;
        }

        // C-layout -> A-layout reshape for P via LDS
        #pragma unroll
        for (int nt = 0; nt < 4; ++nt)
            #pragma unroll
            for (int r = 0; r < 8; ++r)
                Pw[wid][r + ((lane >> 4) << 3)][nt * 16 + (lane & 15)] = f2bf(sf[nt][r]);
        __syncthreads();

        Frag pF0, pF1;
        {
            int m = lane & 15;
            #pragma unroll
            for (int i = 0; i < 16; ++i) {
                int k = ((i >> 3) << 4) + ((lane >> 4) << 3) + (i & 7);
                pF0.u[i] = Pw[wid][m][k];
                pF1.u[i] = Pw[wid][m][32 + k];
            }
        }
        // O += P V (contraction over 64 keys -> two wmma per hd tile)
        #pragma unroll
        for (int nt = 0; nt < 4; ++nt) {
            Frag v0, v1;
            #pragma unroll
            for (int j = 0; j < 16; ++j) {
                int k = ((j >> 3) << 4) + ((lane >> 4) << 3) + (j & 7);
                v0.u[j] = Vt[k][nt * 16 + (lane & 15)];
                v1.u[j] = Vt[32 + k][nt * 16 + (lane & 15)];
            }
            o[nt] = wmma_bf16(pF0, v0, o[nt]);
            o[nt] = wmma_bf16(pF1, v1, o[nt]);
        }
    }

    // normalize and write bf16
    #pragma unroll
    for (int nt = 0; nt < 4; ++nt)
        #pragma unroll
        for (int r = 0; r < 8; ++r) {
            int gm = qrow0 + r + ((lane >> 4) << 3);
            int d  = nt * 16 + (lane & 15);
            out[(size_t)(b * SS + gm) * DM + h * HD + d] = f2bf(o[nt][r] / rs[r]);
        }
}

// ---------------- host-side orchestration ----------------
extern "C" void kernel_launch(void* const* d_in, const int* in_sizes, int n_in,
                              void* d_out, int out_size, void* d_ws, size_t ws_size,
                              hipStream_t stream) {
    const int*   ids    = (const int*)  d_in[0];
    const float* amask  = (const float*)d_in[1];
    const float* wte    = (const float*)d_in[2];
    const float* wpe    = (const float*)d_in[3];
    const float* ln1_g  = (const float*)d_in[4];
    const float* ln1_b  = (const float*)d_in[5];
    const float* qkv_w  = (const float*)d_in[6];
    const float* qkv_b  = (const float*)d_in[7];
    const float* proj_w = (const float*)d_in[8];
    const float* proj_b = (const float*)d_in[9];
    const float* ln2_g  = (const float*)d_in[10];
    const float* ln2_b  = (const float*)d_in[11];
    const float* fc1_w  = (const float*)d_in[12];
    const float* fc1_b  = (const float*)d_in[13];
    const float* fc2_w  = (const float*)d_in[14];
    const float* fc2_b  = (const float*)d_in[15];
    const float* lnf_g  = (const float*)d_in[16];
    const float* lnf_b  = (const float*)d_in[17];
    const float* head_b = (const float*)d_in[18];
    float* logits = (float*)d_out;

    // workspace carve-out: fp32 residual stream + bf16 activations
    char* w = (char*)d_ws;
    float* x            = (float*)w;          w += (size_t)TT * DM * 4;
    unsigned short* lnb = (unsigned short*)w; w += (size_t)TT * DM * 2;
    unsigned short* qkvb= (unsigned short*)w; w += (size_t)TT * D3 * 2;
    unsigned short* attb= (unsigned short*)w; w += (size_t)TT * DM * 2;
    unsigned short* ffh = (unsigned short*)w; w += (size_t)TT * D4 * 2;

    dim3 gQKV (D3  / 128, TT / 128);
    dim3 gPROJ(DM  / 128, TT / 128);
    dim3 gFC1 (D4  / 128, TT / 128);
    dim3 gHEAD(VSZ / 128, TT / 128);
    dim3 gATT (SS / 64, NH, BB);

    embed_kernel<<<TT, 256, 0, stream>>>(ids, wte, wpe, x);

    for (int l = 0; l < NL; ++l) {
        ln_kernel<<<TT, 256, 0, stream>>>(x, ln1_g + l * DM, ln1_b + l * DM, lnb, 1e-6f);
        gemm_kernel<0, false, true><<<gQKV, 256, 0, stream>>>(
            lnb, qkv_w + (size_t)l * DM * D3, qkv_b + (size_t)l * D3,
            nullptr, qkvb, TT, D3, DM);
        attn_kernel<<<gATT, 128, 0, stream>>>(qkvb, amask, attb);
        gemm_kernel<1, false, false><<<gPROJ, 256, 0, stream>>>(
            attb, proj_w + (size_t)l * DM * DM, proj_b + (size_t)l * DM,
            x, x, TT, DM, DM);
        ln_kernel<<<TT, 256, 0, stream>>>(x, ln2_g + l * DM, ln2_b + l * DM, lnb, 1e-6f);
        gemm_kernel<2, false, true><<<gFC1, 256, 0, stream>>>(
            lnb, fc1_w + (size_t)l * DM * D4, fc1_b + (size_t)l * D4,
            nullptr, ffh, TT, D4, DM);
        gemm_kernel<1, false, false><<<gPROJ, 256, 0, stream>>>(
            ffh, fc2_w + (size_t)l * D4 * DM, fc2_b + (size_t)l * DM,
            x, x, TT, DM, D4);
    }

    ln_kernel<<<TT, 256, 0, stream>>>(x, lnf_g, lnf_b, lnb, 1e-5f);
    gemm_kernel<0, true, false><<<gHEAD, 256, 0, stream>>>(
        lnb, wte, head_b, nullptr, logits, TT, VSZ, DM);
}
